// MultiHeadAttention_4131758539490
// MI455X (gfx1250) — compile-verified, matrix operand order checked by reading, not run on hardware
//
#include <hip/hip_runtime.h>
#include <cmath>

typedef __attribute__((ext_vector_type(8)))  float  v8f;
typedef __attribute__((ext_vector_type(16))) __bf16 v16bf;
typedef __attribute__((ext_vector_type(8)))  __bf16 v8bf;

#define DEV_ static __device__ __forceinline__

// ---------- helpers ----------
DEV_ __bf16 f2bf(float f) {
  unsigned u = __float_as_uint(f);
  u += 0x7FFFu + ((u >> 16) & 1u);          // round-to-nearest-even
  unsigned short h = (unsigned short)(u >> 16);
  __bf16 r;
  __builtin_memcpy(&r, &h, 2);
  return r;
}

// A/B fragment load for v_wmma_f32_16x16x32_bf16, row-major 32-element K chunk.
// ISA 16-bit A layout: lanes 0-15 hold K=0..7 (v0-3) and K=16..23 (v4-7);
// lanes 16-31 hold K=8..15 and K=24..31.  half = lane>>4.
DEV_ v16bf load_frag(const __bf16* rowk, int hh) {
  v8bf lo = *reinterpret_cast<const v8bf*>(rowk + 8 * hh);
  v8bf hi = *reinterpret_cast<const v8bf*>(rowk + 16 + 8 * hh);
  return __builtin_shufflevector(lo, hi, 0,1,2,3,4,5,6,7,8,9,10,11,12,13,14,15);
}

DEV_ v8f wmma_bf16(v16bf a, v16bf b, v8f c) {
  return __builtin_amdgcn_wmma_f32_16x16x32_bf16(false, a, false, b, (short)0, c, false, false);
}

// ---------- conversion kernels ----------
__global__ void k_f2bf(const float* __restrict__ x, __bf16* __restrict__ y, int n) {
  int i = blockIdx.x * 256 + threadIdx.x;
  if (i < n) y[i] = f2bf(x[i]);
}

// W [Kd, N] row-major f32  ->  WT [N, Kd] row-major bf16
__global__ void k_transpose_bf(const float* __restrict__ w, __bf16* __restrict__ wt, int Kd, int N) {
  int i = blockIdx.x * 256 + threadIdx.x;
  if (i < Kd * N) {
    int k = i / N, n = i % N;
    wt[(size_t)n * Kd + k] = f2bf(w[i]);
  }
}

// ---------- positional features ----------
__global__ void k_init(unsigned* gmax) {
  if (threadIdx.x == 0 && blockIdx.x == 0) *gmax = 0u;
}

// gamma pdf (double precision) + global max via atomicMax on positive-float bits
__global__ void k_gamma(float* __restrict__ gbuf, unsigned* __restrict__ gmax) {
  int i = blockIdx.x * 256 + threadIdx.x;
  if (i >= 3071 * 32) return;
  int r = i / 32, j = i % 32;
  int px = r - 1535;
  double x = (double)(px < 0 ? -px : px);
  double mean = 48.0 + 1488.0 * (double)j / 31.0;   // linspace(48, 1536, 32)
  double sq = mean / 24.0;                           // stddev = 24
  double conc = sq * sq;
  double rate = mean / 576.0;
  double lu = (conc - 1.0) * log(fmax(x, 1e-300));   // xlogy; x==0 -> exp -> 0
  double ln_ = lgamma(conc) - conc * log(rate);
  float p = (float)exp(lu - ln_) + 1e-8f;
  gbuf[i] = p;
  atomicMax(gmax, __float_as_uint(p));               // p>0: bit pattern monotone
}

// build pe [3072,192] bf16; row 3071 is a zero pad row
__global__ void k_pe(__bf16* __restrict__ peb, const float* __restrict__ gbuf,
                     const unsigned* __restrict__ gmax) {
  int i = blockIdx.x * 256 + threadIdx.x;
  if (i >= 3072 * 192) return;
  int r = i / 192, c = i % 192;
  float val = 0.0f;
  if (r < 3071) {
    int pos = r - 1535;
    float x = fabsf((float)pos);
    int cb = (c < 96) ? c : (c - 96);
    int g = cb >> 5, jf = cb & 31;
    float feat;
    if (g == 0) {                                    // exponential
      float hl = exp2f(3.0f + (10.584962500721156f - 3.0f) * (float)jf / 31.0f);
      feat = exp2f(-x / hl);                         // == exp(-ln2/hl * x)
    } else if (g == 1) {                             // central mask
      float wd = exp2f((float)(jf + 1)) - 1.0f;
      feat = (wd > x) ? 1.0f : 0.0f;
    } else {                                         // gamma, normalized
      feat = gbuf[r * 32 + jf] / __uint_as_float(*gmax);
    }
    val = feat;
    if (c >= 96) {
      float s = (pos > 0) ? 1.0f : ((pos < 0) ? -1.0f : 0.0f);
      val = s * feat;
    }
  }
  peb[i] = f2bf(val);
}

// ---------- epilogue store (shared by GEMM modes) ----------
// MODE 0: Q proj   -> Qc/Qp bf16 [B,H,L,64] with scale 1/8 + biases
// MODE 1: K proj   -> Kb bf16 [B,H,L,64]
// MODE 2: V proj   -> Vt bf16 [B,H,192,L] (transposed)
// MODE 3: rel proj -> RelKb bf16 [H,3072,64]
// MODE 4: out proj -> f32 d_out [3072,1536] + b_emb
template <int MODE>
DEV_ void epi_store(int gM, int gN, float val, float* outF, __bf16* out1, __bf16* out2,
                    const float* bias1, const float* bias2) {
  if (MODE == 0) {
    int hD = gN >> 6, k2 = gN & 63;
    int b = gM / 1536, li = gM - b * 1536;
    size_t idx = ((size_t)((b * 8 + hD) * 1536 + li)) * 64 + k2;
    float v = val * 0.125f;                          // K^-0.5, K=64
    out1[idx] = f2bf(v + bias1[gN]);                 // + rel_content_bias
    out2[idx] = f2bf(v + bias2[gN]);                 // + rel_pos_bias
  } else if (MODE == 1) {
    int hD = gN >> 6, k2 = gN & 63;
    int b = gM / 1536, li = gM - b * 1536;
    out1[((size_t)((b * 8 + hD) * 1536 + li)) * 64 + k2] = f2bf(val);
  } else if (MODE == 2) {
    int hD = gN / 192, n = gN - hD * 192;
    int b = gM / 1536, li = gM - b * 1536;
    out1[((size_t)((b * 8 + hD) * 192 + n)) * 1536 + li] = f2bf(val);
  } else if (MODE == 3) {
    int hD = gN >> 6, k2 = gN & 63;
    out1[((size_t)(hD * 3072 + gM)) * 64 + k2] = f2bf(val);
  } else {
    outF[(size_t)gM * 1536 + gN] = val + bias1[gN];
  }
}

// ---------- WMMA GEMM, 2x4 register blocking: wave owns 32x64 of C ----------
// C[3072,N] = A[3072,Kd](bf16, row-major) x Bt[N,Kd]^T (bf16, pre-transposed).
// Per 32-wide K step: 2 A-frags + 4 B-frags -> 8 WMMAs (loads amortized 2.7x
// vs 1 tile/wave; the WMMA pipe, not VMEM issue, becomes the limiter).
template <int MODE>
__global__ __launch_bounds__(256) void k_gemm(const __bf16* __restrict__ A,
                                              const __bf16* __restrict__ Bt,
                                              int Kd, int nT64,
                                              float* __restrict__ outF,
                                              __bf16* __restrict__ out1,
                                              __bf16* __restrict__ out2,
                                              const float* __restrict__ bias1,
                                              const float* __restrict__ bias2) {
  const int lane = threadIdx.x & 31;
  const int wv = threadIdx.x >> 5;
  const int tile = blockIdx.x * 8 + wv;              // grid sized exactly
  const int mt = tile / nT64, nt = tile % nT64;
  const int m0 = mt * 32, n0 = nt * 64;
  const int hh = lane >> 4, n15 = lane & 15;

  const __bf16* arow0 = A + (size_t)(m0 + n15) * Kd;
  const __bf16* arow1 = A + (size_t)(m0 + 16 + n15) * Kd;
  const __bf16* brow0 = Bt + (size_t)(n0 + n15) * Kd;
  const __bf16* brow1 = Bt + (size_t)(n0 + 16 + n15) * Kd;
  const __bf16* brow2 = Bt + (size_t)(n0 + 32 + n15) * Kd;
  const __bf16* brow3 = Bt + (size_t)(n0 + 48 + n15) * Kd;

  const v8f z8 = {0.f, 0.f, 0.f, 0.f, 0.f, 0.f, 0.f, 0.f};
  v8f acc[2][4];
#pragma unroll
  for (int i = 0; i < 2; ++i)
#pragma unroll
    for (int j = 0; j < 4; ++j) acc[i][j] = z8;

  for (int kk = 0; kk < Kd; kk += 32) {
    v16bf a0 = load_frag(arow0 + kk, hh);
    v16bf a1 = load_frag(arow1 + kk, hh);
    v16bf b0 = load_frag(brow0 + kk, hh);
    v16bf b1 = load_frag(brow1 + kk, hh);
    v16bf b2 = load_frag(brow2 + kk, hh);
    v16bf b3 = load_frag(brow3 + kk, hh);
    acc[0][0] = wmma_bf16(a0, b0, acc[0][0]);
    acc[0][1] = wmma_bf16(a0, b1, acc[0][1]);
    acc[0][2] = wmma_bf16(a0, b2, acc[0][2]);
    acc[0][3] = wmma_bf16(a0, b3, acc[0][3]);
    acc[1][0] = wmma_bf16(a1, b0, acc[1][0]);
    acc[1][1] = wmma_bf16(a1, b1, acc[1][1]);
    acc[1][2] = wmma_bf16(a1, b2, acc[1][2]);
    acc[1][3] = wmma_bf16(a1, b3, acc[1][3]);
  }

#pragma unroll
  for (int i = 0; i < 2; ++i)
#pragma unroll
    for (int j = 0; j < 4; ++j)
#pragma unroll
      for (int r = 0; r < 8; ++r) {
        const int gM = m0 + 16 * i + r + 8 * hh;     // C layout: M = r + 8*half
        const int gN = n0 + 16 * j + n15;
        epi_store<MODE>(gM, gN, acc[i][j][r], outF, out1, out2, bias1, bias2);
      }
}

// ---------- fused relative flash attention ----------
// 1 wave = 16 query rows of one (b,h); 48 column iterations of 32 keys.
// logits[i,j] = Qc_i . K_j  +  Qp_i . relK[j - i + L - 1]
__global__ __launch_bounds__(256) void k_attn(const __bf16* __restrict__ Qc,
                                              const __bf16* __restrict__ Qp,
                                              const __bf16* __restrict__ Kb,
                                              const __bf16* __restrict__ Vt,
                                              const __bf16* __restrict__ RelK,
                                              __bf16* __restrict__ Obf) {
  __shared__ float  ldsPt[8][16][48];
  __shared__ __bf16 ldsP[8][16][32];
  const int lane = threadIdx.x & 31;
  const int w = threadIdx.x >> 5;
  const int wave = blockIdx.x * 8 + w;               // 0..1535
  const int it = wave % 96;
  const int bh = wave / 96;
  const int hD = bh & 7, b = bh >> 3;
  const int i0 = it * 16;
  const int hh = lane >> 4, n15 = lane & 15;

  const __bf16* qrow = Qc + ((size_t)(bh * 1536 + i0 + n15)) * 64;
  const __bf16* prow = Qp + ((size_t)(bh * 1536 + i0 + n15)) * 64;
  v16bf qc0 = load_frag(qrow, hh),      qc1 = load_frag(qrow + 32, hh);
  v16bf qp0 = load_frag(prow, hh),      qp1 = load_frag(prow + 32, hh);

  float mrun[8], lrun[8];
  v8f acc[12];
  const v8f z8 = {0.f, 0.f, 0.f, 0.f, 0.f, 0.f, 0.f, 0.f};
#pragma unroll
  for (int r = 0; r < 8; ++r) { mrun[r] = -3.0e38f; lrun[r] = 0.f; }
#pragma unroll
  for (int t = 0; t < 12; ++t) acc[t] = z8;

  const __bf16* Kbase = Kb + (size_t)bh * 1536 * 64;
  const __bf16* Rbase = RelK + (size_t)hD * 3072 * 64;
  const __bf16* Vbase = Vt + (size_t)bh * 192 * 1536;

  for (int j0 = 0; j0 < 1536; j0 += 32) {
    // ---- content logits: two 16x16 tiles (cols j0..j0+31) ----
    v8f sa = z8, sb = z8;
    {
      const __bf16* kr = Kbase + (size_t)(j0 + n15) * 64;
      sa = wmma_bf16(qc0, load_frag(kr, hh), sa);
      sa = wmma_bf16(qc1, load_frag(kr + 32, hh), sa);
      const __bf16* kr2 = Kbase + (size_t)(j0 + 16 + n15) * 64;
      sb = wmma_bf16(qc0, load_frag(kr2, hh), sb);
      sb = wmma_bf16(qc1, load_frag(kr2 + 32, hh), sb);
    }
    // ---- rel logits: 16x48 band product, gather the Toeplitz diagonal ----
    const int r0 = 1535 + j0 - i0 - 15;              // >= 0; r0+47 <= 3071 (pad row)
#pragma unroll
    for (int msI = 0; msI < 3; ++msI) {
      const int ms = msI * 16;
      const __bf16* rr = Rbase + (size_t)(r0 + ms + n15) * 64;
      v8f pt = z8;
      pt = wmma_bf16(qp0, load_frag(rr, hh), pt);
      pt = wmma_bf16(qp1, load_frag(rr + 32, hh), pt);
#pragma unroll
      for (int r = 0; r < 8; ++r) ldsPt[w][r + 8 * hh][ms + n15] = pt[r];
    }
#pragma unroll
    for (int r = 0; r < 8; ++r) {                    // S2[a,c] = Pt[a, 15 + c - a]
      const int M = r + 8 * hh;
      sa[r] += ldsPt[w][M][15 + n15 - M];
      sb[r] += ldsPt[w][M][31 + n15 - M];
    }
    // ---- online softmax (rows spread over 16 lanes of a half-wave) ----
    float scl[8];
#pragma unroll
    for (int r = 0; r < 8; ++r) {
      float cm = fmaxf(sa[r], sb[r]);
      cm = fmaxf(cm, __shfl_xor(cm, 1, 32));
      cm = fmaxf(cm, __shfl_xor(cm, 2, 32));
      cm = fmaxf(cm, __shfl_xor(cm, 4, 32));
      cm = fmaxf(cm, __shfl_xor(cm, 8, 32));
      const float mn = fmaxf(mrun[r], cm);
      scl[r] = expf(mrun[r] - mn);
      mrun[r] = mn;
      const float pa = expf(sa[r] - mn), pb = expf(sb[r] - mn);
      float rs = pa + pb;
      rs += __shfl_xor(rs, 1, 32);
      rs += __shfl_xor(rs, 2, 32);
      rs += __shfl_xor(rs, 4, 32);
      rs += __shfl_xor(rs, 8, 32);
      lrun[r] = lrun[r] * scl[r] + rs;
      const int M = r + 8 * hh;
      ldsP[w][M][n15]      = f2bf(pa);
      ldsP[w][M][16 + n15] = f2bf(pb);
    }
#pragma unroll
    for (int t = 0; t < 12; ++t)
#pragma unroll
      for (int r = 0; r < 8; ++r) acc[t][r] *= scl[r];
    // ---- rebuild P as A-fragment from LDS (C-layout -> A-layout) ----
    v16bf pf;
#pragma unroll
    for (int j = 0; j < 16; ++j) {
      const int kidx = (j < 8) ? (8 * hh + j) : (16 + 8 * hh + (j - 8));
      pf[j] = ldsP[w][n15][kidx];
    }
    // ---- O += P x V (V stored transposed: contiguous B-fragment loads) ----
#pragma unroll
    for (int t = 0; t < 12; ++t) {
      const __bf16* vr = Vbase + (size_t)(t * 16 + n15) * 1536 + j0;
      acc[t] = wmma_bf16(pf, load_frag(vr, hh), acc[t]);
    }
  }
  // ---- normalize and write O as [B, L, H*192] bf16 for the final GEMM ----
#pragma unroll
  for (int t = 0; t < 12; ++t)
#pragma unroll
    for (int r = 0; r < 8; ++r) {
      const int M = r + 8 * hh;
      const float ov = acc[t][r] / lrun[r];
      const size_t row = (size_t)(b * 1536 + i0 + M);
      Obf[row * 1536 + hD * 192 + t * 16 + n15] = f2bf(ov);
    }
}

// ---------- host launch ----------
extern "C" void kernel_launch(void* const* d_in, const int* in_sizes, int n_in,
                              void* d_out, int out_size, void* d_ws, size_t ws_size,
                              hipStream_t stream) {
  (void)in_sizes; (void)n_in; (void)out_size; (void)ws_size;
  const float* inX  = (const float*)d_in[0];
  const float* Wq   = (const float*)d_in[1];
  const float* Wk   = (const float*)d_in[2];
  const float* Wv   = (const float*)d_in[3];
  const float* Wrel = (const float*)d_in[4];
  const float* Wemb = (const float*)d_in[5];
  const float* bemb = (const float*)d_in[6];
  const float* cbias = (const float*)d_in[7];   // rel_content_bias [H*K]
  const float* pbias = (const float*)d_in[8];   // rel_pos_bias     [H*K]
  float* out = (float*)d_out;

  char* wsp = (char*)d_ws;
  size_t off = 0;
  auto alloc = [&](size_t bytes) -> void* {
    void* p = wsp + off;
    off += (bytes + 255) & ~(size_t)255;
    return p;
  };
  __bf16* Xb    = (__bf16*)alloc(3072ull * 1536 * 2);
  __bf16* WqT   = (__bf16*)alloc(512ull * 1536 * 2);
  __bf16* WkT   = (__bf16*)alloc(512ull * 1536 * 2);
  __bf16* WvT   = (__bf16*)alloc(1536ull * 1536 * 2);
  __bf16* WembT = (__bf16*)alloc(1536ull * 1536 * 2);
  __bf16* WrelT = (__bf16*)alloc(512ull * 192 * 2);
  __bf16* peb   = (__bf16*)alloc(3072ull * 192 * 2);
  __bf16* Qcb   = (__bf16*)alloc(2ull * 8 * 1536 * 64 * 2);
  __bf16* Qpb   = (__bf16*)alloc(2ull * 8 * 1536 * 64 * 2);
  __bf16* Kbb   = (__bf16*)alloc(2ull * 8 * 1536 * 64 * 2);
  __bf16* Vtb   = (__bf16*)alloc(2ull * 8 * 192 * 1536 * 2);
  __bf16* RelKb = (__bf16*)alloc(8ull * 3072 * 64 * 2);
  __bf16* Obf   = (__bf16*)alloc(3072ull * 1536 * 2);
  unsigned* gmax = (unsigned*)alloc(256);
  float* gbuf = (float*)alloc(3071ull * 32 * 4);

  dim3 blk(256);
  k_f2bf<<<(3072 * 1536 + 255) / 256, blk, 0, stream>>>(inX, Xb, 3072 * 1536);
  k_transpose_bf<<<(1536 * 512 + 255) / 256, blk, 0, stream>>>(Wq, WqT, 1536, 512);
  k_transpose_bf<<<(1536 * 512 + 255) / 256, blk, 0, stream>>>(Wk, WkT, 1536, 512);
  k_transpose_bf<<<(1536 * 1536 + 255) / 256, blk, 0, stream>>>(Wv, WvT, 1536, 1536);
  k_transpose_bf<<<(1536 * 1536 + 255) / 256, blk, 0, stream>>>(Wemb, WembT, 1536, 1536);
  k_transpose_bf<<<(192 * 512 + 255) / 256, blk, 0, stream>>>(Wrel, WrelT, 192, 512);

  k_init<<<1, 1, 0, stream>>>(gmax);
  k_gamma<<<(3071 * 32 + 255) / 256, blk, 0, stream>>>(gbuf, gmax);
  k_pe<<<(3072 * 192 + 255) / 256, blk, 0, stream>>>(peb, gbuf, gmax);

  // 3072 rows -> 96 m-tiles of 32; n in 64-wide tiles; grids exactly divisible by 8 waves
  k_gemm<0><<<96, blk, 0, stream>>>(Xb, WqT, 1536, 8, nullptr, Qcb, Qpb, cbias, pbias);
  k_gemm<1><<<96, blk, 0, stream>>>(Xb, WkT, 1536, 8, nullptr, Kbb, nullptr, nullptr, nullptr);
  k_gemm<2><<<288, blk, 0, stream>>>(Xb, WvT, 1536, 24, nullptr, Vtb, nullptr, nullptr, nullptr);
  k_gemm<3><<<96, blk, 0, stream>>>(peb, WrelT, 192, 8, nullptr, RelKb, nullptr, nullptr, nullptr);

  k_attn<<<192, blk, 0, stream>>>(Qcb, Qpb, Kbb, Vtb, RelKb, Obf);

  k_gemm<4><<<288, blk, 0, stream>>>(Obf, WembT, 1536, 24, out, nullptr, nullptr, bemb, nullptr);
}
